// GRUDropout_12025908428829
// MI455X (gfx1250) — compile-verified, matrix operand order checked by reading, not run on hardware
//
#include <hip/hip_runtime.h>
#include <hip/hip_bf16.h>

// GRU (2-layer, variational dropout) for MI455X / gfx1250.
// bf16 WMMA (v_wmma_f32_16x16x32_bf16) GEMMs with fp32 accumulate; one kernel
// launch per timestep (serial recurrence), graph-captured. Input transform is
// fused into the recurrent GEMM (K = D_in + H). x-tile staged into LDS by the
// Tensor Data Mover (tensor_load_to_lds + s_wait_tensorcnt) with TDM-applied
// row padding; h-tile staged by vectorized VALU (dropout + fp32->bf16).

typedef __bf16 bf16_t;
typedef __attribute__((ext_vector_type(16))) __bf16 v16bf;
typedef __attribute__((ext_vector_type(8)))  float  v8f;
typedef __attribute__((ext_vector_type(4)))  unsigned int v4u;
typedef __attribute__((ext_vector_type(8)))  int v8i;
typedef __attribute__((ext_vector_type(4)))  int v4i;

#define P_SCALE 1.25f   // 1/(1-0.2)
#define SB 64           // batch
#define HH 512          // hidden
#define SS 512          // seq len
#define DIN0 256

#if __has_builtin(__builtin_amdgcn_tanhf)
#define TANHF(x) __builtin_amdgcn_tanhf(x)
#else
#define TANHF(x) tanhf(x)
#endif

union V16U {
    v16bf v;
    uint4 q[2];
};

// ---------------------------------------------------------------------------
// Weight packing into WMMA B-tile layout (32x16 bf16 per tile, per-lane
// contiguous 32B). Mapping mirrors the documented 16-bit A-matrix striping:
//   lane L: c = L&15 (column), hi = L>>4
//   elem e: v = e>>1, half = e&1
//   k = kt*32 + (v>=4 ? 16 : 0) + hi*8 + (v&3)*2 + half
//   o = gate*512 + ot*16 + c          (B[k,o] = W[o,k], W row-major [3H,K])
// Packed: [gate(3)][ot(32)][kt(KT)][lane(32)*elem(16)]
// ---------------------------------------------------------------------------
__global__ __launch_bounds__(256) void pack_w_kernel(
    const float* __restrict__ W, bf16_t* __restrict__ P, int K, int KT)
{
    int p = blockIdx.x * 256 + threadIdx.x;
    int total = 3 * 32 * KT * 512;
    if (p >= total) return;
    int idx = p & 511;
    int q = p >> 9;
    int kt = q % KT; q /= KT;
    int ot = q & 31;
    int gate = q >> 5;
    int lane = idx >> 4, e = idx & 15;
    int c = lane & 15, hi = lane >> 4;
    int v = e >> 1, half = e & 1;
    int k = kt * 32 + ((v >= 4) ? 16 : 0) + hi * 8 + (v & 3) * 2 + half;
    int o = gate * 512 + ot * 16 + c;
    P[p] = (bf16_t)W[(size_t)o * K + k];
}

// Pre-apply layer-0 input dropout, store bf16.
__global__ __launch_bounds__(256) void prep_x0_kernel(
    const float* __restrict__ x, const unsigned char* __restrict__ mask_in0,
    bf16_t* __restrict__ xd0, int total)
{
    int i = blockIdx.x * 256 + threadIdx.x;
    if (i >= total) return;
    int bk = i % (SB * DIN0);
    xd0[i] = (bf16_t)(mask_in0[bk] ? 0.f : x[i] * P_SCALE);
}

__global__ __launch_bounds__(256) void zero_f32_kernel(float* __restrict__ p, int n)
{
    int i = blockIdx.x * 256 + threadIdx.x;
    if (i < n) p[i] = 0.f;
}

__device__ __forceinline__ float sigmoidf_(float x) {
    return 1.f / (1.f + __expf(-x));
}

// ---------------------------------------------------------------------------
// One GRU timestep. Grid: 16 blocks x 256 threads (8 wave32) = 128 waves.
// tile = blockIdx.x*8 + wave; mt = tile&3 (16 batch rows), nt = tile>>2
// (16 hidden cols). Each wave: 3 WMMA chains (r/z/n) over K = D_IN + H, with
// separate x-part / h-part accumulators for the n-gate.
// ---------------------------------------------------------------------------
template <int D_IN>
__global__ __launch_bounds__(256) void gru_step_kernel(
    const bf16_t* __restrict__ xd_t,        // [64, D_IN] bf16, dropout applied
    const bf16_t* __restrict__ WihP,        // packed bf16 B-tiles (3*32*KTX*512)
    const bf16_t* __restrict__ WhhP,        // packed bf16 B-tiles (3*32*16*512)
    const float*  __restrict__ b_ih,
    const float*  __restrict__ b_hh,
    const float*  __restrict__ h_prev,      // [64,512] fp32
    const unsigned char* __restrict__ mask_h,
    float* __restrict__ h_next,             // [64,512] fp32
    bf16_t* __restrict__ out_bf16,          // nullable: next-layer dropped input
    const unsigned char* __restrict__ mask_in_next,
    float* __restrict__ out_f32,            // nullable: out1[t]
    float* __restrict__ h_last)             // nullable
{
    constexpr int KTX = D_IN / 32;
    constexpr int KTH = HH / 32;
    constexpr int XSTR = D_IN + 8;          // bf16 elems (row pad keeps b128 align)
    constexpr int HSTR = HH + 8;

    extern __shared__ char smemraw[];
    bf16_t* xs = (bf16_t*)smemraw;          // [64][XSTR] -- filled by TDM
    bf16_t* hs = xs + SB * XSTR;            // [64][HSTR]

    const int tid = threadIdx.x;

    // ---- TDM: async-copy the x tile (64 x D_IN bf16) into LDS, with the
    // row padding (8 bf16 = 4 dwords) applied by the tensor engine itself.
    if (tid < 32) {
        union { v4u v; unsigned int d[4]; } g0 = {};
        unsigned long long ga = (unsigned long long)(uintptr_t)xd_t;
        g0.d[0] = 1u;                               // count = 1 valid descriptor
        g0.d[1] = 0u;                               // lds_addr = 0 (xs @ LDS base)
        g0.d[2] = (unsigned int)ga;                 // global_addr[31:0]
        g0.d[3] = (unsigned int)((ga >> 32) & 0x01FFFFFFu) | (2u << 30); // type=2
        union { v8i v; unsigned int d[8]; } g1 = {};
        constexpr unsigned rows  = SB;
        constexpr unsigned elems = D_IN;
        constexpr unsigned pad_interval = (D_IN == 256) ? 6u : 7u; // 128/256 dw
        g1.d[0] = (1u << 16)                        // data_size = 2 bytes
                | (1u << 20)                        // pad_enable
                | (pad_interval << 22)
                | (3u << 25);                       // pad_amount = 4 dwords
        g1.d[1] = (elems & 0xFFFFu) << 16;          // tensor_dim0 lo16
        g1.d[2] = (elems >> 16) | ((rows & 0xFFFFu) << 16); // td0 hi / td1 lo
        g1.d[3] = (rows >> 16) | ((elems & 0xFFFFu) << 16); // td1 hi / tile_dim0
        g1.d[4] = rows & 0xFFFFu;                   // tile_dim1 (tile_dim2 = 0)
        g1.d[5] = elems;                            // tensor_dim0_stride lo32
        g1.d[6] = 0u;
        g1.d[7] = 0u;
        v4i z4 = {0, 0, 0, 0};
        v8i z8 = {0, 0, 0, 0, 0, 0, 0, 0};
        __builtin_amdgcn_tensor_load_to_lds(g0.v, g1.v, z4, z4, z8, 0);
    }

    // ---- Stage hidden state with dropout, fp32 -> bf16 (vectorized x4),
    // overlapped with the TDM transfer above.
    for (int i4 = tid; i4 < (SB * HH) / 4; i4 += 256) {
        const int i = i4 * 4;
        const int b = i >> 9, j = i & (HH - 1);
        float4 h4 = *(const float4*)(h_prev + i);
        uchar4 m4 = *(const uchar4*)(mask_h + i);
        union { bf16_t e[4]; uint2 u; } o4;
        o4.e[0] = (bf16_t)(m4.x ? 0.f : h4.x * P_SCALE);
        o4.e[1] = (bf16_t)(m4.y ? 0.f : h4.y * P_SCALE);
        o4.e[2] = (bf16_t)(m4.z ? 0.f : h4.z * P_SCALE);
        o4.e[3] = (bf16_t)(m4.w ? 0.f : h4.w * P_SCALE);
        *(uint2*)(hs + b * HSTR + j) = o4.u;
    }

    if (tid < 32) __builtin_amdgcn_s_wait_tensorcnt(0);
    __syncthreads();

    const int wave = tid >> 5;
    const int lane = tid & 31;
    const int tileid = blockIdx.x * 8 + wave;   // 0..127
    const int nt = tileid >> 2;                 // hidden col tile
    const int mt = tileid & 3;                  // batch row tile
    const int c  = lane & 15;
    const int hi = lane >> 4;

    v8f a_r  = {0,0,0,0,0,0,0,0};
    v8f a_z  = {0,0,0,0,0,0,0,0};
    v8f a_in = {0,0,0,0,0,0,0,0};   // x contribution to n-gate
    v8f a_hn = {0,0,0,0,0,0,0,0};   // h contribution to n-gate

    const bf16_t* axrow = xs + (mt * 16 + c) * XSTR + hi * 8;
    const bf16_t* ahrow = hs + (mt * 16 + c) * HSTR + hi * 8;

    // Per-gate per-lane base pointers; all K-tile offsets become immediates.
    const bf16_t* brx = WihP + ((size_t)(0 * 32 + nt) * KTX) * 512 + lane * 16;
    const bf16_t* bzx = WihP + ((size_t)(1 * 32 + nt) * KTX) * 512 + lane * 16;
    const bf16_t* bnx = WihP + ((size_t)(2 * 32 + nt) * KTX) * 512 + lane * 16;
    const bf16_t* brh = WhhP + ((size_t)(0 * 32 + nt) * KTH) * 512 + lane * 16;
    const bf16_t* bzh = WhhP + ((size_t)(1 * 32 + nt) * KTH) * 512 + lane * 16;
    const bf16_t* bnh = WhhP + ((size_t)(2 * 32 + nt) * KTH) * 512 + lane * 16;
    __builtin_prefetch(brx, 0, 1);
    __builtin_prefetch(bzx, 0, 1);
    __builtin_prefetch(bnx, 0, 1);
    __builtin_prefetch(brh, 0, 1);
    __builtin_prefetch(bzh, 0, 1);
    __builtin_prefetch(bnh, 0, 1);

    // ---- x region: accumulate r, z, i_n -----------------------------------
    #pragma unroll
    for (int kt = 0; kt < KTX; ++kt) {
        V16U A, Br, Bz, Bn;
        A.q[0]  = *(const uint4*)(axrow + kt * 32);
        A.q[1]  = *(const uint4*)(axrow + kt * 32 + 16);
        Br.q[0] = *(const uint4*)(brx + kt * 512);
        Br.q[1] = *(const uint4*)(brx + kt * 512 + 8);
        Bz.q[0] = *(const uint4*)(bzx + kt * 512);
        Bz.q[1] = *(const uint4*)(bzx + kt * 512 + 8);
        Bn.q[0] = *(const uint4*)(bnx + kt * 512);
        Bn.q[1] = *(const uint4*)(bnx + kt * 512 + 8);
        a_r  = __builtin_amdgcn_wmma_f32_16x16x32_bf16(false, A.v, false, Br.v,
                                                       (short)0, a_r, false, false);
        a_z  = __builtin_amdgcn_wmma_f32_16x16x32_bf16(false, A.v, false, Bz.v,
                                                       (short)0, a_z, false, false);
        a_in = __builtin_amdgcn_wmma_f32_16x16x32_bf16(false, A.v, false, Bn.v,
                                                       (short)0, a_in, false, false);
    }

    // ---- h region: accumulate r, z, h_n -----------------------------------
    #pragma unroll
    for (int kt = 0; kt < KTH; ++kt) {
        V16U A, Br, Bz, Bn;
        A.q[0]  = *(const uint4*)(ahrow + kt * 32);
        A.q[1]  = *(const uint4*)(ahrow + kt * 32 + 16);
        Br.q[0] = *(const uint4*)(brh + kt * 512);
        Br.q[1] = *(const uint4*)(brh + kt * 512 + 8);
        Bz.q[0] = *(const uint4*)(bzh + kt * 512);
        Bz.q[1] = *(const uint4*)(bzh + kt * 512 + 8);
        Bn.q[0] = *(const uint4*)(bnh + kt * 512);
        Bn.q[1] = *(const uint4*)(bnh + kt * 512 + 8);
        a_r  = __builtin_amdgcn_wmma_f32_16x16x32_bf16(false, A.v, false, Br.v,
                                                       (short)0, a_r, false, false);
        a_z  = __builtin_amdgcn_wmma_f32_16x16x32_bf16(false, A.v, false, Bz.v,
                                                       (short)0, a_z, false, false);
        a_hn = __builtin_amdgcn_wmma_f32_16x16x32_bf16(false, A.v, false, Bn.v,
                                                       (short)0, a_hn, false, false);
    }

    // ---- fused gate epilogue ----------------------------------------------
    // C/D layout: lane -> column N = c (lanes 16-31: M+8), vgpr rr -> row.
    const int o = nt * 16 + c;
    const float br  = b_ih[o]        + b_hh[o];
    const float bz  = b_ih[512 + o]  + b_hh[512 + o];
    const float bin = b_ih[1024 + o];
    const float bhn = b_hh[1024 + o];

    #pragma unroll
    for (int rr = 0; rr < 8; ++rr) {
        const int m = mt * 16 + hi * 8 + rr;      // batch row
        float r = sigmoidf_(a_r[rr] + br);
        float z = sigmoidf_(a_z[rr] + bz);
        float n = TANHF(a_in[rr] + bin + r * (a_hn[rr] + bhn));
        float hd = (float)hs[m * HSTR + o];       // dropped h (as GRUCell sees)
        float hnew = (1.f - z) * n + z * hd;

        const int gidx = m * HH + o;
        h_next[gidx] = hnew;
        if (out_bf16)
            out_bf16[gidx] = (bf16_t)(mask_in_next[gidx] ? 0.f : hnew * P_SCALE);
        if (out_f32)
            out_f32[gidx] = hnew;
        if (h_last)
            h_last[gidx] = hnew;
    }
}

// ---------------------------------------------------------------------------
extern "C" void kernel_launch(void* const* d_in, const int* in_sizes, int n_in,
                              void* d_out, int out_size, void* d_ws, size_t ws_size,
                              hipStream_t stream) {
    (void)in_sizes; (void)n_in; (void)out_size; (void)ws_size;

    const float*         x        = (const float*)d_in[0];
    // jax bool arrays -> 1 byte per element
    const unsigned char* mask_in0 = (const unsigned char*)d_in[1];
    const unsigned char* mask_h0  = (const unsigned char*)d_in[2];
    const unsigned char* mask_in1 = (const unsigned char*)d_in[3];
    const unsigned char* mask_h1  = (const unsigned char*)d_in[4];
    const float* w_ih0 = (const float*)d_in[5];
    const float* w_hh0 = (const float*)d_in[6];
    const float* b_ih0 = (const float*)d_in[7];
    const float* b_hh0 = (const float*)d_in[8];
    const float* w_ih1 = (const float*)d_in[9];
    const float* w_hh1 = (const float*)d_in[10];
    const float* b_ih1 = (const float*)d_in[11];
    const float* b_hh1 = (const float*)d_in[12];

    const int KTx0 = DIN0 / 32;   // 8
    const int KTx1 = HH / 32;     // 16
    const int KTh  = HH / 32;     // 16

    // ---- workspace layout -------------------------------------------------
    char* ws = (char*)d_ws;
    size_t off = 0;
    auto alloc = [&](size_t bytes) -> char* {
        char* p = ws + off;
        off = (off + bytes + 255) & ~(size_t)255;
        return p;
    };
    bf16_t* WihP0 = (bf16_t*)alloc((size_t)3 * 32 * KTx0 * 512 * 2);
    bf16_t* WhhP0 = (bf16_t*)alloc((size_t)3 * 32 * KTh  * 512 * 2);
    bf16_t* WihP1 = (bf16_t*)alloc((size_t)3 * 32 * KTx1 * 512 * 2);
    bf16_t* WhhP1 = (bf16_t*)alloc((size_t)3 * 32 * KTh  * 512 * 2);
    bf16_t* xd0   = (bf16_t*)alloc((size_t)SS * SB * DIN0 * 2);
    bf16_t* xd1   = (bf16_t*)alloc((size_t)SS * SB * HH * 2);
    float*  h0a   = (float*)alloc((size_t)SB * HH * 4);
    float*  h0b   = (float*)alloc((size_t)SB * HH * 4);
    float*  h1a   = (float*)alloc((size_t)SB * HH * 4);
    float*  h1b   = (float*)alloc((size_t)SB * HH * 4);

    float* out1  = (float*)d_out;                       // [S,B,H]
    float* hlast = out1 + (size_t)SS * SB * HH;         // [B,H]

    // ---- prep -------------------------------------------------------------
    {
        int tot = 3 * 32 * KTx0 * 512;
        pack_w_kernel<<<(tot + 255) / 256, 256, 0, stream>>>(w_ih0, WihP0, DIN0, KTx0);
        tot = 3 * 32 * KTh * 512;
        pack_w_kernel<<<(tot + 255) / 256, 256, 0, stream>>>(w_hh0, WhhP0, HH, KTh);
        tot = 3 * 32 * KTx1 * 512;
        pack_w_kernel<<<(tot + 255) / 256, 256, 0, stream>>>(w_ih1, WihP1, HH, KTx1);
        tot = 3 * 32 * KTh * 512;
        pack_w_kernel<<<(tot + 255) / 256, 256, 0, stream>>>(w_hh1, WhhP1, HH, KTh);

        int totx = SS * SB * DIN0;
        prep_x0_kernel<<<(totx + 255) / 256, 256, 0, stream>>>(x, mask_in0, xd0, totx);

        int totz = SB * HH;
        zero_f32_kernel<<<(totz + 255) / 256, 256, 0, stream>>>(h0a, totz);
        zero_f32_kernel<<<(totz + 255) / 256, 256, 0, stream>>>(h1a, totz);
    }

    // ---- layer 0 recurrence (writes xd1 = dropped layer-1 input) ----------
    const size_t smem0 = (size_t)(SB * (DIN0 + 8) + SB * (HH + 8)) * 2;
    for (int t = 0; t < SS; ++t) {
        const bf16_t* xt = xd0 + (size_t)t * SB * DIN0;
        float* hp = (t & 1) ? h0b : h0a;
        float* hn = (t & 1) ? h0a : h0b;
        gru_step_kernel<DIN0><<<16, 256, smem0, stream>>>(
            xt, WihP0, WhhP0, b_ih0, b_hh0, hp, mask_h0, hn,
            xd1 + (size_t)t * SB * HH, mask_in1,
            nullptr, nullptr);
    }

    // ---- layer 1 recurrence (writes out1 and h_last) ----------------------
    const size_t smem1 = (size_t)(SB * (HH + 8) + SB * (HH + 8)) * 2;
    for (int t = 0; t < SS; ++t) {
        const bf16_t* xt = xd1 + (size_t)t * SB * HH;
        float* hp = (t & 1) ? h1b : h1a;
        float* hn = (t & 1) ? h1a : h1b;
        gru_step_kernel<HH><<<16, 256, smem1, stream>>>(
            xt, WihP1, WhhP1, b_ih1, b_hh1, hp, mask_h1, hn,
            nullptr, nullptr,
            out1 + (size_t)t * SB * HH,
            (t == SS - 1) ? hlast : nullptr);
    }
}
